// LSTM1997Layer_71219147702764
// MI455X (gfx1250) — matrix operand order repaired and checked by reading, not compiled, not run on hardware
//
#include <hip/hip_runtime.h>
#include <math.h>

// ---------------------------------------------------------------------------
// LSTM-1997 for MI455X (gfx1250, wave32, WMMA 16x16x32 bf16).
//   Phase 1: Xpre[m][0:1152] = x[m][:] @ [Wi;Wo;Wg]^T + [bi;bo;bg]  (WMMA GEMM)
//   Phase 2: for t in 0..S-1 (stream-serialized):
//     hproj:  pre_part[kc] = h_{t-1} @ Ucat^T  (K split 4 ways, WMMA)
//     cell :  i,o=sigmoid, g=tanh, c_t, h_t    (elementwise, reduces kc)
// bf16 operands, f32 accumulation. Weight tiles staged to LDS with
// global_load_async_to_lds_b128 (ASYNCcnt); activations converted f32->bf16
// through VALU (v_cvt_pk_bf16_f32) with vectorized b128/b64 traffic.
// ---------------------------------------------------------------------------

typedef __bf16 bf16;
typedef __attribute__((ext_vector_type(4)))  bf16  bfx4;
typedef __attribute__((ext_vector_type(8)))  bf16  bfx8;
typedef __attribute__((ext_vector_type(16))) bf16  bfx16;
typedef __attribute__((ext_vector_type(4)))  float fx4;
typedef __attribute__((ext_vector_type(8)))  float fx8;

#define B_    32
#define S_    1024
#define K_    1024      // H_IN
#define HOUT  1024
#define NT    1152      // 64 (i) + 64 (o) + 1024 (g)
#define LDST  40        // LDS row stride in bf16: 80B, keeps 16B chunk alignment

static __device__ __forceinline__ bfx16 cat8(bfx8 lo, bfx8 hi) {
  bfx16 r;
#pragma unroll
  for (int i = 0; i < 8; ++i) { r[i] = lo[i]; r[i + 8] = hi[i]; }
  return r;
}

static __device__ __forceinline__ float sigmoidf_(float x) {
  return 1.0f / (1.0f + __expf(-x));
}

static __device__ __forceinline__ unsigned lds_off(const void* p) {
  // generic -> LDS(AS3) byte offset for async-to-LDS VDST operand
  return (unsigned)(size_t)(__attribute__((address_space(3))) const void*)p;
}

static __device__ __forceinline__ void async_tile16(unsigned loff,
                                                    const void* gsrc) {
  unsigned long long ga = (unsigned long long)(size_t)gsrc;
  asm volatile("global_load_async_to_lds_b128 %0, %1, off"
               :: "v"(loff), "v"(ga) : "memory");
}

static __device__ __forceinline__ void wait_async0() {
  asm volatile("s_wait_asynccnt 0" ::: "memory");
}

static __device__ __forceinline__ bfx4 cvt4(fx4 v) {
  bfx4 b;
  b[0] = (bf16)v[0]; b[1] = (bf16)v[1]; b[2] = (bf16)v[2]; b[3] = (bf16)v[3];
  return b;
}

// --------------------------- weight/bias packing ---------------------------
__global__ void lstm_pack_kernel(const float* __restrict__ Wi, const float* __restrict__ Wo,
                                 const float* __restrict__ Wg, const float* __restrict__ Ui,
                                 const float* __restrict__ Uo, const float* __restrict__ Ug,
                                 const float* __restrict__ bi, const float* __restrict__ bo,
                                 const float* __restrict__ bg,
                                 bf16* __restrict__ Wcat, bf16* __restrict__ Ucat,
                                 float* __restrict__ bcat) {
  int idx = blockIdx.x * blockDim.x + threadIdx.x;        // 0 .. NT*K_-1
  int n = idx >> 10, k = idx & (K_ - 1);
  float w, u;
  if (n < 64)       { w = Wi[n * K_ + k];         u = Ui[n * K_ + k]; }
  else if (n < 128) { w = Wo[(n - 64) * K_ + k];  u = Uo[(n - 64) * K_ + k]; }
  else              { w = Wg[(n - 128) * K_ + k]; u = Ug[(n - 128) * K_ + k]; }
  Wcat[idx] = (bf16)w;
  Ucat[idx] = (bf16)u;
  if (idx < NT) {
    float bv = (idx < 64) ? bi[idx] : (idx < 128) ? bo[idx - 64] : bg[idx - 128];
    bcat[idx] = bv;
  }
}

// ------------------- phase 1: input projections (big GEMM) -----------------
// Block: 256 thr (8 waves). Tile: 128(M) x 64(N). Wave: 32x32 = 2x2 WMMA tiles.
__global__ void lstm_xproj_kernel(const float* __restrict__ x,
                                  const bf16* __restrict__ Wcat,
                                  const float* __restrict__ bcat,
                                  float* __restrict__ Xpre) {
  __shared__ __align__(16) bf16 xs[128 * LDST];
  __shared__ __align__(16) bf16 wsm[64 * LDST];

  const int tid   = threadIdx.x;
  const int lane  = tid & 31, wave = tid >> 5;
  const int half  = lane >> 4, l15 = lane & 15;
  const int waveM = wave >> 1, waveN = wave & 1;       // 4 x 2 waves
  const int n_base = blockIdx.x * 64;
  const int m_base = blockIdx.y * 128;

  fx8 acc[2][2] = {};

  for (int k0 = 0; k0 < K_; k0 += 32) {
    // W tile: 64x32 bf16, async global->LDS, one 16B chunk per thread
    {
      int r = tid >> 2, c8 = (tid & 3) * 8;
      async_tile16(lds_off(&wsm[r * LDST + c8]),
                   &Wcat[(size_t)(n_base + r) * K_ + (k0 + c8)]);
    }
    // x tile: 128x32 f32 -> bf16 (float4 loads, packed cvt, b64 LDS stores)
#pragma unroll
    for (int it = 0; it < 4; ++it) {
      int id = it * 256 + tid;                 // 0..1023 float4 chunk id
      int r = id >> 3, c = (id & 7) * 4;
      fx4 v = *(const fx4*)&x[(size_t)(m_base + r) * K_ + (k0 + c)];
      *(bfx4*)&xs[r * LDST + c] = cvt4(v);
    }
    if (k0 + 32 < K_)  // pull next x K-tile toward L2 (global_prefetch_b8)
      __builtin_prefetch(&x[(size_t)(m_base + (tid >> 1)) * K_ + (k0 + 32) + (tid & 1) * 16], 0, 1);
    wait_async0();
    __syncthreads();

    // A fragments (16-bit A 16x32 layout: lane-half picks K base 0/8, halves at +16)
    bfx16 afrag[2], bfrag[2];
#pragma unroll
    for (int mi = 0; mi < 2; ++mi) {
      int rowA = waveM * 32 + mi * 16 + l15;
      afrag[mi] = cat8(*(const bfx8*)&xs[rowA * LDST + half * 8],
                       *(const bfx8*)&xs[rowA * LDST + 16 + half * 8]);
    }
    // B fragments (same K layout as A: lane-half picks K base 0/8, halves at +16)
#pragma unroll
    for (int ni = 0; ni < 2; ++ni) {
      int rowB = waveN * 32 + ni * 16 + l15;
      bfrag[ni] = cat8(*(const bfx8*)&wsm[rowB * LDST + half * 8],
                       *(const bfx8*)&wsm[rowB * LDST + 16 + half * 8]);
    }
#pragma unroll
    for (int mi = 0; mi < 2; ++mi)
#pragma unroll
      for (int ni = 0; ni < 2; ++ni)
        acc[mi][ni] = __builtin_amdgcn_wmma_f32_16x16x32_bf16(
            false, afrag[mi], false, bfrag[ni], (short)0, acc[mi][ni], false, false);
    __syncthreads();
  }

  // epilogue: D element r maps to M = r + 8*half; N = l15
#pragma unroll
  for (int mi = 0; mi < 2; ++mi)
#pragma unroll
    for (int ni = 0; ni < 2; ++ni) {
      int n = n_base + waveN * 32 + ni * 16 + l15;
      float bias = bcat[n];
#pragma unroll
      for (int r = 0; r < 8; ++r) {
        int m = m_base + waveM * 32 + mi * 16 + r + half * 8;
        Xpre[(size_t)m * NT + n] = acc[mi][ni][r] + bias;
      }
    }
}

// ---------------- phase 2a: recurrent projection (K-split GEMM) ------------
// Grid: (NT/64, 4 K-chunks). Block: 256 thr (8 waves), 32(M) x 64(N) tile,
// one 16x16 WMMA tile per wave. Partial sums to pre_part[kc][32][NT].
__global__ void lstm_hproj_kernel(const float* __restrict__ hprev, long hstride,
                                  const bf16* __restrict__ Ucat,
                                  float* __restrict__ pre_part) {
  __shared__ __align__(16) bf16 hs[32 * LDST];
  __shared__ __align__(16) bf16 us[64 * LDST];

  const int tid  = threadIdx.x;
  const int lane = tid & 31, wave = tid >> 5;
  const int half = lane >> 4, l15 = lane & 15;
  const int mi = (wave & 1) * 16;        // 2 (M) x 4 (N) waves
  const int ni = (wave >> 1) * 16;
  const int n_base = blockIdx.x * 64;
  const int kc = blockIdx.y;             // K chunk of 256

  fx8 acc = {};

  for (int kk = 0; kk < 256; kk += 32) {
    const int k0 = kc * 256 + kk;
    // U tile: 64x32 bf16, async global->LDS, one 16B chunk per thread
    {
      int r = tid >> 2, c8 = (tid & 3) * 8;
      async_tile16(lds_off(&us[r * LDST + c8]),
                   &Ucat[(size_t)(n_base + r) * K_ + (k0 + c8)]);
    }
    // h tile: 32x32 f32 -> bf16, one float4 per thread
    {
      int r = tid >> 3, c = (tid & 7) * 4;
      fx4 v = *(const fx4*)&hprev[(size_t)r * hstride + (k0 + c)];
      *(bfx4*)&hs[r * LDST + c] = cvt4(v);
    }
    wait_async0();
    __syncthreads();

    int rowA = mi + l15;
    bfx16 a = cat8(*(const bfx8*)&hs[rowA * LDST + half * 8],
                   *(const bfx8*)&hs[rowA * LDST + 16 + half * 8]);
    int rowB = ni + l15;
    bfx16 b = cat8(*(const bfx8*)&us[rowB * LDST + half * 8],
                   *(const bfx8*)&us[rowB * LDST + 16 + half * 8]);
    acc = __builtin_amdgcn_wmma_f32_16x16x32_bf16(false, a, false, b, (short)0, acc,
                                                  false, false);
    __syncthreads();
  }

#pragma unroll
  for (int r = 0; r < 8; ++r) {
    int m_local = mi + r + half * 8;               // 0..31 (batch row)
    int n = n_base + ni + l15;
    pre_part[((size_t)kc * 32 + m_local) * NT + n] = acc[r];
  }
}

// ----------------------- phase 2b: cell elementwise ------------------------
__global__ void lstm_cell_kernel(const float* __restrict__ Xpre,
                                 const float* __restrict__ pre_part,
                                 const float* __restrict__ c0,
                                 float* __restrict__ c_out,
                                 float* __restrict__ h_out, int t) {
  int idx = blockIdx.x * blockDim.x + threadIdx.x;   // 0 .. B_*HOUT-1
  int b  = idx >> 10;
  int hh = idx & 1023;
  int nb = hh >> 4;                                  // block index (64 blocks of 16)
  size_t m = (size_t)b * S_ + t;

  const float* xp = Xpre + m * NT;
  float pi = 0.f, po = 0.f, pg = 0.f;
#pragma unroll
  for (int kc = 0; kc < 4; ++kc) {
    const float* pp = pre_part + ((size_t)kc * 32 + b) * NT;
    pi += pp[nb];
    po += pp[64 + nb];
    pg += pp[128 + hh];
  }
  float i_t = sigmoidf_(xp[nb] + pi);
  float o_t = sigmoidf_(xp[64 + nb] + po);
  float g_t = tanhf(xp[128 + hh] + pg);

  float c_prev = (t == 0) ? c0[b * HOUT + hh] : c_out[(m - 1) * HOUT + hh];
  float c_new  = c_prev + i_t * g_t;
  float h_new  = o_t * tanhf(c_new);

  c_out[m * HOUT + hh] = c_new;
  h_out[m * HOUT + hh] = h_new;
}

// --------------------------------- host ------------------------------------
extern "C" void kernel_launch(void* const* d_in, const int* in_sizes, int n_in,
                              void* d_out, int out_size, void* d_ws, size_t ws_size,
                              hipStream_t stream) {
  (void)in_sizes; (void)n_in; (void)out_size; (void)ws_size;
  const float* x  = (const float*)d_in[0];
  const float* h0 = (const float*)d_in[1];
  const float* c0 = (const float*)d_in[2];
  const float* Wi = (const float*)d_in[3];
  const float* bi = (const float*)d_in[4];
  const float* Wo = (const float*)d_in[5];
  const float* bo = (const float*)d_in[6];
  const float* Wg = (const float*)d_in[7];
  const float* bg = (const float*)d_in[8];
  const float* Ui = (const float*)d_in[9];
  const float* Uo = (const float*)d_in[10];
  const float* Ug = (const float*)d_in[11];

  float* c_out = (float*)d_out;                       // (B,S,64,16) flat
  float* h_out = c_out + (size_t)B_ * S_ * HOUT;      // (B,S,1024) flat

  // workspace layout
  char*  ws   = (char*)d_ws;
  size_t off  = 0;
  float* Xpre = (float*)(ws + off); off += (size_t)B_ * S_ * NT * sizeof(float);
  bf16*  Wcat = (bf16*)(ws + off);  off += (size_t)NT * K_ * sizeof(bf16);
  bf16*  Ucat = (bf16*)(ws + off);  off += (size_t)NT * K_ * sizeof(bf16);
  float* bcat = (float*)(ws + off); off += (size_t)NT * sizeof(float);
  off = (off + 255) & ~(size_t)255;
  float* pre_part = (float*)(ws + off);               // [4][32][NT]

  lstm_pack_kernel<<<(NT * K_) / 256, 256, 0, stream>>>(Wi, Wo, Wg, Ui, Uo, Ug,
                                                        bi, bo, bg, Wcat, Ucat, bcat);
  lstm_xproj_kernel<<<dim3(NT / 64, (B_ * S_) / 128), 256, 0, stream>>>(x, Wcat, bcat, Xpre);

  for (int t = 0; t < S_; ++t) {
    const float* hprev = (t == 0) ? h0 : (h_out + (size_t)(t - 1) * HOUT);
    long hstride = (t == 0) ? (long)HOUT : (long)S_ * HOUT;
    lstm_hproj_kernel<<<dim3(NT / 64, 4), 256, 0, stream>>>(hprev, hstride, Ucat, pre_part);
    lstm_cell_kernel<<<(B_ * HOUT) / 256, 256, 0, stream>>>(Xpre, pre_part, c0,
                                                            c_out, h_out, t);
  }
}
